// GCNSpatialBlock_18476949307589
// MI455X (gfx1250) — matrix operand order repaired and checked by reading, not compiled, not run on hardware
//
#include <hip/hip_runtime.h>
#include <hip/hip_bf16.h>

// ---------------------------------------------------------------------------
// GCN spatial block, fused for MI455X (gfx1250, wave32, WMMA).
//   out[bt,n,o] = diag[n]*M[n,o]*(x@W0)[bt,n,o]
//              + sum_m a_off[n,m]*M[m,o]*(x@W1)[bt,m,o] + bias[o], all * 1e-9
// Roofline: 334 MB HBM traffic @ 23.3 TB/s ~= 14.3 us -> memory bound.
// GEMMs as bf16x3-split V_WMMA_F32_16X16X32_BF16 (~13 us compute, near-fp32
// accuracy), fused with 17x17 node mixing through LDS. This revision trades
// register pressure for occupancy: two weight passes keep <=8 B-fragments
// live (~150 VGPRs, no >256-VGPR MSB addressing) and 75 KB LDS allows
// 4 workgroups (36 waves) per WGP for HBM latency hiding.
// ---------------------------------------------------------------------------

typedef __attribute__((ext_vector_type(16))) __bf16 v16bf;
typedef __attribute__((ext_vector_type(8)))  float  v8f;

#define NUM_NODES 17
#define DIM 64
#define TOK_PER_WG 8                          // 8*17 = 136 rows
#define ROWS_PER_WG (TOK_PER_WG * NUM_NODES)  // 136
#define TILES_PER_WG 9                        // ceil(136/16), last tile padded
#define ROWS_PAD (TILES_PER_WG * 16)          // 144
#define NTHREADS 288                          // 9 waves, 1 row-tile each

// ---- workspace layout (bytes) ----
#define WS_AMAT   0                          // 289 f32  (symmetrized off-diag)
#define WS_DM     1280                       // 1088 f32 (diag[n]*M[n][o])
#define WS_WT0HI  5632                       // 4096 bf16, W0^T hi  [col][k]
#define WS_WT0LO  13824                      // 4096 bf16, W0^T lo
#define WS_WT1HI  22016                      // 4096 bf16, W1^T hi
#define WS_WT1LO  30208                      // 4096 bf16, W1^T lo

// ---------------------------------------------------------------------------
// Prep kernel (one block): fold adjacency, diag*M, and split/transpose W.
// ---------------------------------------------------------------------------
__global__ __launch_bounds__(256) void gcn_prep_kernel(
    const float* __restrict__ W,    // [2,64,64]
    const float* __restrict__ M,    // [17,64]
    const float* __restrict__ adj2, // [17,17]
    const float* __restrict__ adj,  // [17,17]
    float* __restrict__ ws) {
  __shared__ float diag_s[NUM_NODES];
  float*  amat = (float*)((char*)ws + WS_AMAT);
  float*  dM   = (float*)((char*)ws + WS_DM);
  __bf16* wt0h = (__bf16*)((char*)ws + WS_WT0HI);
  __bf16* wt0l = (__bf16*)((char*)ws + WS_WT0LO);
  __bf16* wt1h = (__bf16*)((char*)ws + WS_WT1HI);
  __bf16* wt1l = (__bf16*)((char*)ws + WS_WT1LO);

  int t = threadIdx.x;
  if (t < NUM_NODES * NUM_NODES) {
    int n = t / NUM_NODES, m = t % NUM_NODES;
    float a_nm = adj[n * NUM_NODES + m] + adj2[n * NUM_NODES + m];
    float a_mn = adj[m * NUM_NODES + n] + adj2[m * NUM_NODES + n];
    float s = 0.5f * (a_nm + a_mn);
    amat[t] = (n == m) ? 0.0f : s;
    if (n == m) diag_s[n] = s;
  }
  __syncthreads();
  for (int i = t; i < NUM_NODES * DIM; i += 256) {
    int n = i / DIM;
    dM[i] = diag_s[n] * M[i];
  }
  // W split: store transposed [col][k] so B-fragment K pairs are contiguous.
  for (int i = t; i < 2 * DIM * DIM; i += 256) {
    int c   = i / (DIM * DIM);
    int rem = i - c * DIM * DIM;
    int col = rem / DIM;
    int k   = rem % DIM;
    float w  = W[c * DIM * DIM + k * DIM + col];
    __bf16 hi = (__bf16)w;
    __bf16 lo = (__bf16)(w - (float)hi);
    if (c == 0) { wt0h[col * DIM + k] = hi; wt0l[col * DIM + k] = lo; }
    else        { wt1h[col * DIM + k] = hi; wt1l[col * DIM + k] = lo; }
  }
}

// ---------------------------------------------------------------------------
// Fused main kernel: 1 WG = 8 tokens (136 rows -> 9 padded row-tiles),
// 288 threads = 9 waves, one 16-row WMMA tile per wave per pass.
// ---------------------------------------------------------------------------
__global__ __launch_bounds__(NTHREADS) void gcn_fused_kernel(
    const float* __restrict__ x,     // [totalRows, 64]
    const float* __restrict__ Mg,    // [17,64]
    const float* __restrict__ biasg, // [64]
    const float* __restrict__ ws,
    float* __restrict__ out,
    int totalRows) {
  extern __shared__ float lds[];
  float* P     = lds;                     // [144][64] diag*M*h0 (padded)
  float* G1    = P + ROWS_PAD * DIM;      // [144][64] M*h1      (padded)
  float* sAmat = G1 + ROWS_PAD * DIM;     // [17][17]
  float* sBias = sAmat + NUM_NODES * NUM_NODES; // [64]

  const float*  amat = (const float*)((const char*)ws + WS_AMAT);
  const float*  dM   = (const float*)((const char*)ws + WS_DM);
  const __bf16* wt0h = (const __bf16*)((const char*)ws + WS_WT0HI);
  const __bf16* wt0l = (const __bf16*)((const char*)ws + WS_WT0LO);
  const __bf16* wt1h = (const __bf16*)((const char*)ws + WS_WT1HI);
  const __bf16* wt1l = (const __bf16*)((const char*)ws + WS_WT1LO);

  for (int i = threadIdx.x; i < NUM_NODES * NUM_NODES; i += NTHREADS)
    sAmat[i] = amat[i];
  if (threadIdx.x < DIM) sBias[threadIdx.x] = biasg[threadIdx.x];
  __syncthreads();

  const int wave = threadIdx.x >> 5;     // 0..8 == row-tile index
  const int lane = threadIdx.x & 31;
  const int l16  = lane & 15;
  const int half = lane >> 4;
  const size_t rowBase = (size_t)blockIdx.x * ROWS_PER_WG;
  const int r0 = wave * 16;

  // Clamp A-row for the padded 9th tile of the last workgroup (results of
  // padded rows land in the LDS pad region and are never read).
  size_t rg = rowBase + r0 + l16;
  if (rg >= (size_t)totalRows) rg = (size_t)totalRows - 1;
  const float* xrow = x + rg * DIM;

  // ---- stage 1: two passes (W0 -> P, W1 -> G1), bf16x3-split WMMA ----
#pragma clang loop unroll(disable)
  for (int pass = 0; pass < 2; ++pass) {
    const __bf16* wh  = pass ? wt1h : wt0h;
    const __bf16* wl  = pass ? wt1l : wt0l;
    const float*  scl = pass ? Mg   : dM;     // [17][64] elementwise scale
    float*        dst = pass ? G1   : P;

    // A fragments (16x32 bf16), hi/lo split; rebuilt per pass (L0-resident).
    v16bf aHi[2], aLo[2];
#pragma unroll
    for (int kc = 0; kc < 2; ++kc) {
#pragma unroll
      for (int v = 0; v < 8; ++v) {
        int kb = kc * 32 + ((v < 4) ? (half * 8 + 2 * v) : (16 + half * 8 + 2 * (v - 4)));
        float f0 = xrow[kb], f1 = xrow[kb + 1];
        __bf16 h0 = (__bf16)f0, h1 = (__bf16)f1;
        aHi[kc][2 * v]     = h0;
        aHi[kc][2 * v + 1] = h1;
        aLo[kc][2 * v]     = (__bf16)(f0 - (float)h0);
        aLo[kc][2 * v + 1] = (__bf16)(f1 - (float)h1);
      }
    }
#pragma unroll
    for (int j = 0; j < 4; ++j) {
      const int col  = j * 16 + l16;
      const int colk = col * DIM;
      v16bf bh[2], bl[2];
#pragma unroll
      for (int kc = 0; kc < 2; ++kc) {
#pragma unroll
        for (int v = 0; v < 8; ++v) {
          int k = kc * 32 + ((v < 4) ? (half * 8 + 2 * v) : (16 + half * 8 + 2 * (v - 4)));
          bh[kc][2 * v] = wh[colk + k]; bh[kc][2 * v + 1] = wh[colk + k + 1];
          bl[kc][2 * v] = wl[colk + k]; bl[kc][2 * v + 1] = wl[colk + k + 1];
        }
      }
      v8f acc = {};
#pragma unroll
      for (int kc = 0; kc < 2; ++kc) {
        acc = __builtin_amdgcn_wmma_f32_16x16x32_bf16(false, aHi[kc], false, bh[kc], (short)0, acc, false, false);
        acc = __builtin_amdgcn_wmma_f32_16x16x32_bf16(false, aHi[kc], false, bl[kc], (short)0, acc, false, false);
        acc = __builtin_amdgcn_wmma_f32_16x16x32_bf16(false, aLo[kc], false, bh[kc], (short)0, acc, false, false);
      }
      // C/D layout: VGPR v -> row v + 8*half, col = lane&15.
#pragma unroll
      for (int v = 0; v < 8; ++v) {
        int rl = r0 + v + 8 * half;          // < 144 (pad region ok)
        int n  = rl % NUM_NODES;
        dst[rl * DIM + col] = scl[n * DIM + col] * acc[v];
      }
    }
  }
  __syncthreads();

  // ---- stage 2: dense 17x17 node mixing + bias, write out (coalesced) ----
  for (int i = threadIdx.x; i < ROWS_PER_WG * DIM; i += NTHREADS) {
    int rl  = i >> 6;
    int c   = i & 63;
    int tok = rl / NUM_NODES;
    int n   = rl - tok * NUM_NODES;
    const float* g  = G1 + tok * NUM_NODES * DIM + c;
    const float* ar = sAmat + n * NUM_NODES;
    float s = P[i] + sBias[c];
#pragma unroll
    for (int m = 0; m < NUM_NODES; ++m) s += ar[m] * g[m * DIM];
    out[rowBase * DIM + i] = s * 1e-9f;
  }
}

extern "C" void kernel_launch(void* const* d_in, const int* in_sizes, int n_in,
                              void* d_out, int out_size, void* d_ws, size_t ws_size,
                              hipStream_t stream) {
  const float* x    = (const float*)d_in[0];
  const float* W    = (const float*)d_in[1];
  const float* M    = (const float*)d_in[2];
  const float* adj2 = (const float*)d_in[3];
  const float* bias = (const float*)d_in[4];
  const float* adj  = (const float*)d_in[5];
  float* out = (float*)d_out;

  int totalRows = in_sizes[0] / DIM;            // B*T*17 = 652800
  int numTokens = totalRows / NUM_NODES;        // B*T    = 38400
  int numGroups = numTokens / TOK_PER_WG;       // 4800

  gcn_prep_kernel<<<1, 256, 0, stream>>>(W, M, adj2, adj, (float*)d_ws);

  size_t ldsBytes =
      (size_t)(2 * ROWS_PAD * DIM +             // P, G1 (padded to 144 rows)
               NUM_NODES * NUM_NODES + DIM)     // amat, bias
      * sizeof(float);                          // ~75 KB -> 4 WGs per WGP

  gcn_fused_kernel<<<numGroups, NTHREADS, ldsBytes, stream>>>(
      x, M, bias, (const float*)d_ws, out, totalRows);
}
